// MultiHeadAttention_45612552684065
// MI455X (gfx1250) — matrix operand order repaired and checked by reading, not compiled, not run on hardware
//
#include <hip/hip_runtime.h>
#include <hip/hip_bf16.h>

// MHA forward for MI455X (gfx1250): bf16 WMMA matmuls, fp32 softmax/accum.
// B=2, S=2048, E=768, H=12, D=64.
#define EMBED 768
#define HEADS 12
#define HDIM  64
#define SEQ   2048
#define BATCH 2
#define NROW  (BATCH*SEQ)       // 4096 token rows
#define KT_PROJ (EMBED/32)      // 24 k-steps of 32
#define SC_LD 2052              // padded LDS score-row stride (floats)
#define LDA   40                // LDS tile row stride in bf16 (32 data + 8 pad, 80B: 16B-aligned rows, bank spread)

typedef __attribute__((ext_vector_type(16))) __bf16 v16bf;
typedef __attribute__((ext_vector_type(8)))  __bf16 v8bf;
typedef __attribute__((ext_vector_type(8)))  float  v8f;
typedef __attribute__((ext_vector_type(4)))  float  v4f;

// ---- WMMA wrapper: D = A(16x32 bf16) * B(32x16 bf16) + C(16x16 f32) ----
static __device__ __forceinline__ v8f wmma_bf16(v16bf a, v16bf b, v8f c) {
  return __builtin_amdgcn_wmma_f32_16x16x32_bf16(false, a, false, b, (short)0, c,
                                                 false, false);
}

// ---- CDNA5 async global->LDS copy (16B) + wait (ASYNCcnt) ----
static __device__ __forceinline__ void async_cp16(const __bf16* gsrc, const __bf16* lds) {
  unsigned ldsOff = (unsigned)(unsigned long long)(const void*)lds;  // addr[31:0] = LDS offset
  asm volatile("global_load_async_to_lds_b128 %0, %1, off"
               :: "v"(ldsOff), "v"((unsigned long long)gsrc)
               : "memory");
}
static __device__ __forceinline__ void wait_async0() {
  asm volatile("s_wait_asynccnt 0" ::: "memory");
}

// A-operand (16x32 bf16): lane-lo holds K[0..15], lane-hi K[16..31].
// `base` already includes +hi*16 within a 32-element k-row.
static __device__ __forceinline__ v16bf ldA_bf16(const __bf16* base) {
  v8bf lo = *(const v8bf*)base;
  v8bf hb = *(const v8bf*)(base + 8);
  v16bf r;
#pragma unroll
  for (int i = 0; i < 8; ++i) { r[i] = lo[i]; r[8 + i] = hb[i]; }
  return r;
}

// B-operand: 16 contiguous bf16 (lane = column N, K-half chosen by caller).
// Two 16B loads so only 16B alignment is required.
static __device__ __forceinline__ v16bf ldB_bf16(const __bf16* p) {
  v8bf p0 = *(const v8bf*)p;
  v8bf p1 = *(const v8bf*)(p + 8);
  v16bf r;
#pragma unroll
  for (int i = 0; i < 8; ++i) { r[i] = p0[i]; r[8 + i] = p1[i]; }
  return r;
}

// A-operand from fp32 (LDS score strip): same split-chunk pattern.
static __device__ __forceinline__ v16bf cvtA_f32(const float* base) {
  const v4f* q = (const v4f*)base;
  v4f x0 = q[0], x1 = q[1];   // K 0..7
  v4f x2 = q[2], x3 = q[3];   // K 8..15
  v16bf r;
#pragma unroll
  for (int i = 0; i < 4; ++i) {
    r[i]      = (__bf16)x0[i];
    r[4 + i]  = (__bf16)x1[i];
    r[8 + i]  = (__bf16)x2[i];
    r[12 + i] = (__bf16)x3[i];
  }
  return r;
}

// ============================================================================
// Kernel 0: one-pass fp32 -> bf16 conversion of X (q,k,v) and W (Wq,Wk,Wv,Wo)
// into a single concatenated bf16 buffer. 8 elements / thread.
// ============================================================================
__global__ __launch_bounds__(256) void mha_cvt_inputs(
    const float* __restrict__ q, const float* __restrict__ k,
    const float* __restrict__ v,
    const float* __restrict__ Wq, const float* __restrict__ Wk,
    const float* __restrict__ Wv, const float* __restrict__ Wo,
    __bf16* __restrict__ dst) {
  const size_t XE = (size_t)NROW * EMBED;   // 3,145,728
  const size_t WE = (size_t)EMBED * EMBED;  //   589,824
  size_t g8 = ((size_t)blockIdx.x * 256 + threadIdx.x) * 8;
  const float* src; size_t off;
  if      (g8 <   XE)          { src = q;  off = g8; }
  else if (g8 < 2*XE)          { src = k;  off = g8 -   XE; }
  else if (g8 < 3*XE)          { src = v;  off = g8 - 2*XE; }
  else if (g8 < 3*XE +   WE)   { src = Wq; off = g8 - 3*XE; }
  else if (g8 < 3*XE + 2*WE)   { src = Wk; off = g8 - 3*XE -   WE; }
  else if (g8 < 3*XE + 3*WE)   { src = Wv; off = g8 - 3*XE - 2*WE; }
  else                         { src = Wo; off = g8 - 3*XE - 3*WE; }
  v4f x0 = *(const v4f*)(src + off);
  v4f x1 = *(const v4f*)(src + off + 4);
  v8bf o;
#pragma unroll
  for (int i = 0; i < 4; ++i) { o[i] = (__bf16)x0[i]; o[4 + i] = (__bf16)x1[i]; }
  *(v8bf*)(dst + g8) = o;
}

// ---- shared staging helper: 128x32 bf16 A tile + 128x32 bf16 B tile ----
// 512 16B-chunks per tile; 256 threads -> 2 chunks each per tile.
static __device__ __forceinline__ void stage_tiles(
    const __bf16* Asrc, const __bf16* Bsrc,  // already at (row0, k0)
    __bf16* Ab, __bf16* Bb, int tid) {
#pragma unroll
  for (int cc = 0; cc < 2; ++cc) {
    int c = tid * 2 + cc;            // 0..511
    int row = c >> 2, seg = c & 3;   // 4 x 16B per 32-elem row
    async_cp16(Asrc + (size_t)row * EMBED + seg * 8, Ab + row * LDA + seg * 8);
    async_cp16(Bsrc + (size_t)row * EMBED + seg * 8, Bb + row * LDA + seg * 8);
  }
}

// ============================================================================
// Kernel 1: Q/K/V projection, LDS-blocked GEMM. y = x*W^T + b, bf16 out.
// Block = 128x128 output tile; 8 waves in 4(M)x2(N); wave = 32x64 = 8 WMMA acc.
// K staged 32-deep, double-buffered via async global->LDS copies.
// Q,K -> [B,H,S,D]; V -> transposed [B,H,D,S].
// ============================================================================
__global__ __launch_bounds__(256) void mha_proj_gemm(
    const __bf16* __restrict__ Xb, const __bf16* __restrict__ Wb,
    const float* __restrict__ bq, const float* __restrict__ bk,
    const float* __restrict__ bv,
    __bf16* __restrict__ Qo, __bf16* __restrict__ Ko, __bf16* __restrict__ Vt) {
  __shared__ __bf16 smA[2][128 * LDA];
  __shared__ __bf16 smB[2][128 * LDA];
  const int tid = threadIdx.x;
  const int lane = tid & 31, wave = tid >> 5;
  const int m = lane & 15, hi = lane >> 4;
  const int waveM = wave & 3, waveN = wave >> 2;

  int blk = blockIdx.x;            // 3 mats * 32 mtiles * 6 ntiles = 576
  int mat = blk / 192;
  int rem = blk % 192;
  int m0 = (rem / 6) * 128;
  int n0 = (rem % 6) * 128;

  const __bf16* Asrc = Xb + (size_t)mat * ((size_t)NROW * EMBED) + (size_t)m0 * EMBED;
  const __bf16* Bsrc = Wb + (size_t)mat * ((size_t)EMBED * EMBED) + (size_t)n0 * EMBED;
  const float* bias = (mat == 0) ? bq : (mat == 1) ? bk : bv;

  v8f acc[8] = {};  // [msub 0..1][nsub 0..3]

  stage_tiles(Asrc, Bsrc, smA[0], smB[0], tid);
  wait_async0();
  __syncthreads();

  for (int kt = 0; kt < KT_PROJ; ++kt) {
    int cur = kt & 1;
    if (kt + 1 < KT_PROJ)
      stage_tiles(Asrc + (kt + 1) * 32, Bsrc + (kt + 1) * 32,
                  smA[cur ^ 1], smB[cur ^ 1], tid);

    const __bf16* Ab = smA[cur];
    const __bf16* Bb = smB[cur];
    v16bf a0 = ldA_bf16(Ab + (waveM * 32 + m) * LDA + hi * 16);
    v16bf a1 = ldA_bf16(Ab + (waveM * 32 + 16 + m) * LDA + hi * 16);
    v16bf b0 = ldB_bf16(Bb + (waveN * 64 +  0 + m) * LDA + hi * 16);
    v16bf b1 = ldB_bf16(Bb + (waveN * 64 + 16 + m) * LDA + hi * 16);
    v16bf b2 = ldB_bf16(Bb + (waveN * 64 + 32 + m) * LDA + hi * 16);
    v16bf b3 = ldB_bf16(Bb + (waveN * 64 + 48 + m) * LDA + hi * 16);
    acc[0] = wmma_bf16(a0, b0, acc[0]);
    acc[1] = wmma_bf16(a0, b1, acc[1]);
    acc[2] = wmma_bf16(a0, b2, acc[2]);
    acc[3] = wmma_bf16(a0, b3, acc[3]);
    acc[4] = wmma_bf16(a1, b0, acc[4]);
    acc[5] = wmma_bf16(a1, b1, acc[5]);
    acc[6] = wmma_bf16(a1, b2, acc[6]);
    acc[7] = wmma_bf16(a1, b3, acc[7]);

    wait_async0();
    __syncthreads();
  }

  // Epilogue: bias + bf16 scatter into head-major Q/K or transposed V.
#pragma unroll
  for (int nsub = 0; nsub < 4; ++nsub) {
    int col = n0 + waveN * 64 + nsub * 16 + m;
    float bval = bias[col];
    int h = col >> 6, d = col & 63;
#pragma unroll
    for (int msub = 0; msub < 2; ++msub) {
      v8f a = acc[msub * 4 + nsub];
#pragma unroll
      for (int r = 0; r < 8; ++r) {
        int g = m0 + waveM * 32 + msub * 16 + r + hi * 8;
        int bidx = g >> 11, s = g & (SEQ - 1);
        __bf16 val = (__bf16)(a[r] + bval);
        if (mat == 2)
          Vt[(((size_t)bidx * HEADS + h) * HDIM + d) * SEQ + s] = val;
        else if (mat == 1)
          Ko[(((size_t)bidx * HEADS + h) * SEQ + s) * HDIM + d] = val;
        else
          Qo[(((size_t)bidx * HEADS + h) * SEQ + s) * HDIM + d] = val;
      }
    }
  }
}

// ============================================================================
// Kernel 2: attention. One block = (b,h, 16 q-rows), 8 waves.
//  A: S = Q K^T / 8 via WMMA -> 16x2048 fp32 LDS strip (K/V have no
//     intra-block reuse; cross-block reuse lives in the 192MB L2).
//  B: softmax (shuffle reductions), coalesced fp32 attn write, normalize LDS.
//  C: O = P V via WMMA, k-split across wave pairs + LDS reduction.
// ============================================================================
__global__ __launch_bounds__(256) void mha_attn(
    const __bf16* __restrict__ Qb, const __bf16* __restrict__ Kb,
    const __bf16* __restrict__ Vt, float* __restrict__ attn,
    __bf16* __restrict__ ctx) {
  extern __shared__ char smem[];
  float* sc   = (float*)smem;                                 // [16][SC_LD]
  float* part = (float*)(smem + 16 * SC_LD * sizeof(float));  // [4][16][16]

  const int lane = threadIdx.x & 31;
  const int wave = threadIdx.x >> 5;
  const int m = lane & 15, hi = lane >> 4;

  const int bh = blockIdx.x >> 7;
  const int q0 = (blockIdx.x & 127) * 16;

  const __bf16* qr = Qb + ((size_t)bh * SEQ + q0 + m) * HDIM;
  v16bf a0 = ldA_bf16(qr + hi * 16);
  v16bf a1 = ldA_bf16(qr + 32 + hi * 16);

  // ---- Phase A: scores ----
  for (int t = wave; t < SEQ / 16; t += 8) {
    int kb = t * 16;
    const __bf16* kr = Kb + ((size_t)bh * SEQ + kb + m) * HDIM;
    __builtin_prefetch((const void*)(kr + 128 * HDIM), 0, 3);
    v16bf b0 = *(const v16bf*)(kr + hi * 16);
    v16bf b1 = *(const v16bf*)(kr + 32 + hi * 16);
    v8f acc = {};
    acc = wmma_bf16(a0, b0, acc);
    acc = wmma_bf16(a1, b1, acc);
#pragma unroll
    for (int r = 0; r < 8; ++r)
      sc[(r + hi * 8) * SC_LD + kb + m] = acc[r] * 0.125f;  // 1/sqrt(64)
  }
  __syncthreads();

  // ---- Phase B: softmax, 2 rows per wave ----
  for (int rr = 0; rr < 2; ++rr) {
    int row = wave * 2 + rr;
    float* srow = sc + row * SC_LD;
    float mx = -3.402823466e38f;
    for (int c = lane; c < SEQ; c += 32) mx = fmaxf(mx, srow[c]);
#pragma unroll
    for (int off = 16; off > 0; off >>= 1) mx = fmaxf(mx, __shfl_xor(mx, off, 32));
    float sum = 0.f;
    for (int c = lane; c < SEQ; c += 32) {
      float e = __expf(srow[c] - mx);
      srow[c] = e;
      sum += e;
    }
#pragma unroll
    for (int off = 16; off > 0; off >>= 1) sum += __shfl_xor(sum, off, 32);
    float inv = 1.f / sum;
    float* arow = attn + ((size_t)bh * SEQ + q0 + row) * SEQ;
    for (int c = lane; c < SEQ; c += 32) {
      float p = srow[c] * inv;
      srow[c] = p;
      arow[c] = p;
    }
  }
  __syncthreads();

  // ---- Phase C: O = P * V ----
  const int d0 = (wave & 3) * 16;
  const int kh = (wave >> 2) * (SEQ / 2);
  const __bf16* vr = Vt + ((size_t)bh * HDIM + d0 + m) * SEQ;
  v8f acc = {};
  for (int kk = 0; kk < SEQ / 2; kk += 32) {
    int kb = kh + kk;
    v16bf a  = cvtA_f32(sc + (size_t)m * SC_LD + kb + hi * 16);
    v16bf bv = *(const v16bf*)(vr + kb + hi * 16);
    acc = wmma_bf16(a, bv, acc);
  }
  if (wave >= 4) {
#pragma unroll
    for (int r = 0; r < 8; ++r)
      part[((wave & 3) * 16 + r + hi * 8) * 16 + m] = acc[r];
  }
  __syncthreads();
  if (wave < 4) {
    int bidx = bh / HEADS, h = bh % HEADS;
#pragma unroll
    for (int r = 0; r < 8; ++r) {
      int row = r + hi * 8;
      float val = acc[r] + part[((wave & 3) * 16 + row) * 16 + m];
      ctx[((size_t)bidx * SEQ + q0 + row) * EMBED + h * HDIM + d0 + m] = (__bf16)val;
    }
  }
}

// ============================================================================
// Kernel 3: output projection, same LDS-blocked GEMM; fp32 out + bias.
// ============================================================================
__global__ __launch_bounds__(256) void mha_out_gemm(
    const __bf16* __restrict__ ctx, const __bf16* __restrict__ Wob,
    const float* __restrict__ bo, float* __restrict__ out) {
  __shared__ __bf16 smA[2][128 * LDA];
  __shared__ __bf16 smB[2][128 * LDA];
  const int tid = threadIdx.x;
  const int lane = tid & 31, wave = tid >> 5;
  const int m = lane & 15, hi = lane >> 4;
  const int waveM = wave & 3, waveN = wave >> 2;

  int m0 = (blockIdx.x / 6) * 128;   // 32 * 6 = 192 blocks
  int n0 = (blockIdx.x % 6) * 128;

  const __bf16* Asrc = ctx + (size_t)m0 * EMBED;
  const __bf16* Bsrc = Wob + (size_t)n0 * EMBED;

  v8f acc[8] = {};
  stage_tiles(Asrc, Bsrc, smA[0], smB[0], tid);
  wait_async0();
  __syncthreads();

  for (int kt = 0; kt < KT_PROJ; ++kt) {
    int cur = kt & 1;
    if (kt + 1 < KT_PROJ)
      stage_tiles(Asrc + (kt + 1) * 32, Bsrc + (kt + 1) * 32,
                  smA[cur ^ 1], smB[cur ^ 1], tid);
    const __bf16* Ab = smA[cur];
    const __bf16* Bb = smB[cur];
    v16bf a0 = ldA_bf16(Ab + (waveM * 32 + m) * LDA + hi * 16);
    v16bf a1 = ldA_bf16(Ab + (waveM * 32 + 16 + m) * LDA + hi * 16);
    v16bf b0 = ldB_bf16(Bb + (waveN * 64 +  0 + m) * LDA + hi * 16);
    v16bf b1 = ldB_bf16(Bb + (waveN * 64 + 16 + m) * LDA + hi * 16);
    v16bf b2 = ldB_bf16(Bb + (waveN * 64 + 32 + m) * LDA + hi * 16);
    v16bf b3 = ldB_bf16(Bb + (waveN * 64 + 48 + m) * LDA + hi * 16);
    acc[0] = wmma_bf16(a0, b0, acc[0]);
    acc[1] = wmma_bf16(a0, b1, acc[1]);
    acc[2] = wmma_bf16(a0, b2, acc[2]);
    acc[3] = wmma_bf16(a0, b3, acc[3]);
    acc[4] = wmma_bf16(a1, b0, acc[4]);
    acc[5] = wmma_bf16(a1, b1, acc[5]);
    acc[6] = wmma_bf16(a1, b2, acc[6]);
    acc[7] = wmma_bf16(a1, b3, acc[7]);
    wait_async0();
    __syncthreads();
  }

#pragma unroll
  for (int nsub = 0; nsub < 4; ++nsub) {
    int col = n0 + waveN * 64 + nsub * 16 + m;
    float bval = bo[col];
#pragma unroll
    for (int msub = 0; msub < 2; ++msub) {
      v8f a = acc[msub * 4 + nsub];
#pragma unroll
      for (int r = 0; r < 8; ++r) {
        int g = m0 + waveM * 32 + msub * 16 + r + hi * 8;
        out[(size_t)g * EMBED + col] = a[r] + bval;
      }
    }
  }
}

// ============================================================================
extern "C" void kernel_launch(void* const* d_in, const int* in_sizes, int n_in,
                              void* d_out, int out_size, void* d_ws, size_t ws_size,
                              hipStream_t stream) {
  (void)in_sizes; (void)n_in; (void)out_size; (void)ws_size;
  const float* q  = (const float*)d_in[0];
  const float* k  = (const float*)d_in[1];
  const float* v  = (const float*)d_in[2];
  const float* Wq = (const float*)d_in[3];
  const float* bq = (const float*)d_in[4];
  const float* Wk = (const float*)d_in[5];
  const float* bk = (const float*)d_in[6];
  const float* Wv = (const float*)d_in[7];
  const float* bv = (const float*)d_in[8];
  const float* Wo = (const float*)d_in[9];
  const float* bo = (const float*)d_in[10];

  // Workspace: [ XWb bf16 (q|k|v|Wq|Wk|Wv|Wo) | Qb | Kb | Vt | ctx ]  ~48.8 MB
  char* ws = (char*)d_ws;
  const size_t XE = (size_t)NROW * EMBED;
  const size_t WE = (size_t)EMBED * EMBED;
  const size_t xwBytes  = (3 * XE + 4 * WE) * sizeof(__bf16);   // 23,592,960
  const size_t qkvBytes = (size_t)BATCH * HEADS * SEQ * HDIM * sizeof(__bf16);
  __bf16* XWb = (__bf16*)ws;
  __bf16* Qb  = (__bf16*)(ws + xwBytes);
  __bf16* Kb  = (__bf16*)(ws + xwBytes + qkvBytes);
  __bf16* Vt  = (__bf16*)(ws + xwBytes + 2 * qkvBytes);
  __bf16* ctx = (__bf16*)(ws + xwBytes + 3 * qkvBytes);
  __bf16* Wb  = XWb + 3 * XE;     // weight region
  __bf16* Wob = Wb + 3 * WE;      // Wo bf16

  float* out  = (float*)d_out;                       // [B,S,E]
  float* attn = out + (size_t)BATCH * SEQ * EMBED;   // [B,H,S,S]

  const size_t smemAttn = 16 * SC_LD * sizeof(float) + 4 * 16 * 16 * sizeof(float);
  (void)hipFuncSetAttribute((const void*)mha_attn,
                            hipFuncAttributeMaxDynamicSharedMemorySize,
                            (int)smemAttn);

  mha_cvt_inputs<<<5760, 256, 0, stream>>>(q, k, v, Wq, Wk, Wv, Wo, XWb);
  mha_proj_gemm<<<576, 256, 0, stream>>>(XWb, Wb, bq, bk, bv, Qb, Kb, Vt);
  mha_attn<<<BATCH * HEADS * (SEQ / 16), 256, smemAttn, stream>>>(Qb, Kb, Vt, attn, ctx);
  mha_out_gemm<<<192, 256, 0, stream>>>(ctx, Wob, bo, out);
}
